// ENNModelWithSparsityControl_37641093382748
// MI455X (gfx1250) — matrix operand order repaired and checked by reading, not compiled, not run on hardware
//
#include <hip/hip_runtime.h>
#include <hip/hip_bf16.h>

// ---------------------------------------------------------------------------
// ENN model, MI455X (gfx1250) implementation.
// Dominant cost: 3x GEMM  Y[b,m,s] = sum_n x[b,n,s]*mixing[n,m]
//   (M=8192, K=8192, 512 columns) -> bf16 WMMA with f32 accumulate.
// mixing is transposed+converted to bf16 ONCE (128 MiB, fits 192 MB L2).
// GEMM K-tiles are double-buffered in LDS and staged with inline-asm
// GLOBAL_LOAD_ASYNC_TO_LDS_B128 (tracked on ASYNCcnt, overlapped with WMMA).
// ---------------------------------------------------------------------------

typedef __attribute__((ext_vector_type(16))) __bf16 v16bf;
typedef __attribute__((ext_vector_type(8)))  __bf16 v8bf;
typedef __attribute__((ext_vector_type(8)))  float  v8f;

#define N_  8192
#define B_  8
#define S_  64
#define C_  16

#define MT  128   // GEMM rows per workgroup
#define KT  64    // K tile
#define LDA 72    // LDS row stride (halves), padded vs 64 banks
#define LDB 72

static __device__ __forceinline__ v16bf cat16(v8bf lo, v8bf hi) {
  v16bf r;
#pragma unroll
  for (int i = 0; i < 8; ++i) { r[i] = lo[i]; r[i + 8] = hi[i]; }
  return r;
}

// 32-bit LDS byte address of a __shared__ pointer (AS(3) repr is the LDS offset)
static __device__ __forceinline__ unsigned ldsAddr(const void* p) {
  return (unsigned)(unsigned long long)(__attribute__((address_space(3))) const char*)p;
}

// GLOBAL_LOAD_ASYNC_TO_LDS_B128, GV mode: vdst = LDS byte addr, vaddr = 64b VA
static __device__ __forceinline__ void asyncLoadB128(unsigned lds_a, const void* g) {
  asm volatile("global_load_async_to_lds_b128 %0, %1, off"
               :: "v"(lds_a), "v"(g) : "memory");
}

static __device__ __forceinline__ void wait_async0() {
#if __has_builtin(__builtin_amdgcn_s_wait_asynccnt)
  __builtin_amdgcn_s_wait_asynccnt(0);
#else
  asm volatile("s_wait_asynccnt 0x0" ::: "memory");
#endif
}

// --------------------------------------------------------------------------
// 1) At[m][k] = (bf16) mixing[k][m]   (tiled 64x64 transpose via LDS)
// --------------------------------------------------------------------------
__global__ __launch_bounds__(256) void prep_mixing_kernel(
    const float* __restrict__ mix, __bf16* __restrict__ At) {
  __shared__ float tile[64][65];
  const int t  = threadIdx.x;
  const int m0 = blockIdx.x * 64;
  const int k0 = blockIdx.y * 64;
#pragma unroll
  for (int i = 0; i < 16; ++i) {            // load: rows = k, cols = m (coalesced)
    int lin = t + i * 256;
    int kl = lin >> 6, ml = lin & 63;
    tile[kl][ml] = mix[(size_t)(k0 + kl) * N_ + (m0 + ml)];
  }
  __syncthreads();
#pragma unroll
  for (int i = 0; i < 16; ++i) {            // store: rows = m, cols = k (coalesced)
    int lin = t + i * 256;
    int ml = lin >> 6, kl = lin & 63;
    At[(size_t)(m0 + ml) * N_ + (k0 + kl)] = (__bf16)tile[kl][ml];
  }
}

// --------------------------------------------------------------------------
// 2) xgT[b][s][n] = (bf16)( x[b][n][s] * sigmoid(ent[n][s]) )
// --------------------------------------------------------------------------
__global__ __launch_bounds__(256) void gate_xpose_kernel(
    const float* __restrict__ x, const float* __restrict__ ent,
    __bf16* __restrict__ xgT) {
  __shared__ float tile[64][65];
  const int t  = threadIdx.x;
  const int n0 = blockIdx.x * 64;
  const int b  = blockIdx.y;
#pragma unroll
  for (int i = 0; i < 16; ++i) {
    int lin = t + i * 256;
    int nl = lin >> 6, s = lin & 63;
    float e = ent[(size_t)(n0 + nl) * S_ + s];
    float g = 1.0f / (1.0f + expf(-e));
    tile[nl][s] = x[((size_t)b * N_ + (n0 + nl)) * S_ + s] * g;
  }
  __syncthreads();
#pragma unroll
  for (int i = 0; i < 16; ++i) {
    int lin = t + i * 256;
    int sl = lin >> 6, nl = lin & 63;
    xgT[((size_t)b * S_ + sl) * N_ + (n0 + nl)] = (__bf16)tile[nl][sl];
  }
}

// --------------------------------------------------------------------------
// 3) WMMA GEMM: Y[b][m][s] = sum_k At[m][k] * xgT[b][s][k]
//    Block: 256 threads (8 waves). Tile 128(m) x 64(s), K-tile 64, LDS
//    double-buffered. Wave (w&3, w>>2) owns a 32x32 quadrant -> 2x2 accums.
// --------------------------------------------------------------------------
__global__ __launch_bounds__(256) void gemm_mix_kernel(
    const __bf16* __restrict__ At,    // [N][N]  (row m, col k)
    const __bf16* __restrict__ xgT,   // [B][S][N] (row s, col k)
    float* __restrict__ Y) {          // [B][N][S]
  __shared__ __bf16 ldsA[2][MT * LDA];   // [m][k]
  __shared__ __bf16 ldsB[2][S_ * LDB];   // [s][k]
  const int t    = threadIdx.x;
  const int wave = t >> 5;
  const int lane = t & 31;
  const int col  = lane & 15;         // N index inside tile / A row
  const int rsel = lane >> 4;         // half-select per ISA lane layout
  const int m0   = blockIdx.x * MT;
  const int b    = blockIdx.y;
  const int wmi  = (wave & 3) * 32;   // wave row quadrant
  const int wni  = (wave >> 2) * 32;  // wave col quadrant

  const __bf16* Abase = At  + (size_t)m0 * N_;
  const __bf16* Bbase = xgT + (size_t)b * (S_ * N_);

  // per-thread staging assignment: A = 4x b128, B = 2x b128 per K-tile
  int rowA[4], u4A[4], rowB[2], u4B[2];
#pragma unroll
  for (int i = 0; i < 4; ++i) { int lin = t + i * 256; rowA[i] = lin >> 3; u4A[i] = lin & 7; }
#pragma unroll
  for (int i = 0; i < 2; ++i) { int lin = t + i * 256; rowB[i] = lin >> 3; u4B[i] = lin & 7; }

  // precomputed per-thread LDS destinations (byte addresses) for each buffer
  unsigned ldsDstA[2][4], ldsDstB[2][2];
#pragma unroll
  for (int bu = 0; bu < 2; ++bu) {
#pragma unroll
    for (int i = 0; i < 4; ++i)
      ldsDstA[bu][i] = ldsAddr(&ldsA[bu][rowA[i] * LDA + u4A[i] * 8]);
#pragma unroll
    for (int i = 0; i < 2; ++i)
      ldsDstB[bu][i] = ldsAddr(&ldsB[bu][rowB[i] * LDB + u4B[i] * 8]);
  }

#define STAGE_ISSUE(bufidx, k0v)                                                    \
  do {                                                                              \
    _Pragma("unroll")                                                               \
    for (int i_ = 0; i_ < 4; ++i_)                                                  \
      asyncLoadB128(ldsDstA[bufidx][i_],                                            \
                    Abase + (size_t)rowA[i_] * N_ + (k0v) + u4A[i_] * 8);           \
    _Pragma("unroll")                                                               \
    for (int i_ = 0; i_ < 2; ++i_)                                                  \
      asyncLoadB128(ldsDstB[bufidx][i_],                                            \
                    Bbase + (size_t)rowB[i_] * N_ + (k0v) + u4B[i_] * 8);           \
  } while (0)

  v8f acc[2][2] = {{{}, {}}, {{}, {}}};

  // prologue: fill buffer 0
  STAGE_ISSUE(0, 0);
  wait_async0();
  __syncthreads();

  for (int k0 = 0; k0 < N_; k0 += KT) {
    const int cur = (k0 >> 6) & 1;
    const int nxt = cur ^ 1;
    const bool more = (k0 + KT) < N_;
    if (more) STAGE_ISSUE(nxt, k0 + KT);   // DMA overlapped with compute below

#pragma unroll
    for (int kk = 0; kk < KT; kk += 32) {
      v16bf af[2], bfr[2];
      // A fragment (16x32): lanes 0-15 -> M=lane, K {0..7,16..23};
      // lanes 16-31 -> M=lane-16, K {8..15,24..31}
#pragma unroll
      for (int i = 0; i < 2; ++i) {
        const __bf16* pa = &ldsA[cur][(wmi + i * 16 + col) * LDA + kk + rsel * 8];
        af[i] = cat16(*(const v8bf*)pa, *(const v8bf*)(pa + 16));
        // B fragment (32x16): lanes 0-15 -> N=lane, K {0..7,16..23};
        // lanes 16-31 -> N=lane-16, K {8..15,24..31}
        const __bf16* pb = &ldsB[cur][(wni + i * 16 + col) * LDB + kk + rsel * 8];
        bfr[i] = cat16(*(const v8bf*)pb, *(const v8bf*)(pb + 16));
      }
#pragma unroll
      for (int i = 0; i < 2; ++i)
#pragma unroll
        for (int j = 0; j < 2; ++j)
          acc[i][j] = __builtin_amdgcn_wmma_f32_16x16x32_bf16(
              false, af[i], false, bfr[j], (short)0, acc[i][j], false, false);
    }

    if (more) wait_async0();   // own next-tile DMAs done
    __syncthreads();           // everyone done reading cur & writing nxt
  }

  // Epilogue: C/D layout -> VGPR r holds (M = r + rsel*8, N = col)
#pragma unroll
  for (int i = 0; i < 2; ++i) {
    float* Yb = Y + ((size_t)b * N_ + m0 + wmi + i * 16 + rsel * 8) * S_;
#pragma unroll
    for (int j = 0; j < 2; ++j)
#pragma unroll
      for (int r = 0; r < 8; ++r)
        Yb[(size_t)r * S_ + wni + j * 16 + col] = acc[i][j][r];
  }
#undef STAGE_ISSUE
}

// --------------------------------------------------------------------------
// 4) slot[m][s] = mean_b Y[b][m][s]
// --------------------------------------------------------------------------
__global__ __launch_bounds__(256) void mean_kernel(
    const float* __restrict__ Y, float* __restrict__ slot) {
  size_t idx = (size_t)blockIdx.x * 256 + threadIdx.x;   // over N*S
  float a = 0.f;
#pragma unroll
  for (int b = 0; b < B_; ++b) a += Y[(size_t)b * (N_ * S_) + idx];
  slot[idx] = a * (1.0f / B_);
}

// --------------------------------------------------------------------------
// 5) Final ns: temporal mix -> enc(ReLU) -> dec -> top-8(|.|) collapse
// --------------------------------------------------------------------------
__global__ __launch_bounds__(64) void final_ns_kernel(
    const float* __restrict__ s0, const float* __restrict__ s1,
    const float* __restrict__ s2,
    const float* __restrict__ encw, const float* __restrict__ encb,
    const float* __restrict__ decw, const float* __restrict__ decb,
    float* __restrict__ nsout, float w0, float w1, float w2) {
  __shared__ float nsrow[S_];
  __shared__ float h[C_];
  __shared__ float absv[S_];
  __shared__ float thr;
  const int m = blockIdx.x, s = threadIdx.x;
  const size_t idx = (size_t)m * S_ + s;
  nsrow[s] = w0 * s0[idx] + w1 * s1[idx] + w2 * s2[idx];
  __syncthreads();
  if (s < C_) {
    float a = encb[s];
#pragma unroll
    for (int k = 0; k < S_; ++k) a += nsrow[k] * encw[s * S_ + k];
    h[s] = fmaxf(a, 0.f);
  }
  __syncthreads();
  float v = decb[s];
#pragma unroll
  for (int c = 0; c < C_; ++c) v += h[c] * decw[s * C_ + c];
  float a = fabsf(v);
  absv[s] = a;
  __syncthreads();
  if (s == 0) {                       // 8th largest (duplicates counted, like top_k)
    float kth = 0.f;
    for (int k = 0; k < 8; ++k) {
      float mx = -1.f; int mi = 0;
      for (int j = 0; j < S_; ++j)
        if (absv[j] > mx) { mx = absv[j]; mi = j; }
      absv[mi] = -1.f;
      kth = mx;
    }
    thr = kth;
  }
  __syncthreads();
  nsout[idx] = (a >= thr) ? v : 0.f;
}

// --------------------------------------------------------------------------
// 6) out[b,n,s] = sum_s' x[b,n,s'] * readout_w[s][s']   (f32, memory bound)
// --------------------------------------------------------------------------
__global__ __launch_bounds__(256) void readout_kernel(
    const float* __restrict__ x, const float* __restrict__ rw,
    float* __restrict__ out) {
  __shared__ float wl[S_][65];
  __shared__ float xr[4][S_];
  const int t = threadIdx.x;
  const int rloc = t >> 6, s = t & 63;
#pragma unroll
  for (int i = 0; i < 16; ++i) {
    int lin = t + i * 256;
    wl[lin >> 6][lin & 63] = rw[lin];
  }
  const size_t row = (size_t)blockIdx.x * 4 + rloc;   // over B*N rows
  xr[rloc][s] = x[row * S_ + s];
  __syncthreads();
  float a = 0.f;
#pragma unroll
  for (int k = 0; k < S_; ++k) a += xr[rloc][k] * wl[s][k];
  out[row * S_ + s] = a;
}

// --------------------------------------------------------------------------
// Host-side orchestration.  Workspace layout (needs ~158 MiB):
//   [0,128Mi)        bf16 At = mixing^T
//   [128Mi,136Mi)    bf16 xgT
//   [136Mi,152Mi)    f32 x ping buffer
//   [152Mi,158Mi)    3x f32 mean slots
// --------------------------------------------------------------------------
extern "C" void kernel_launch(void* const* d_in, const int* in_sizes, int n_in,
                              void* d_out, int out_size, void* d_ws, size_t ws_size,
                              hipStream_t stream) {
  (void)in_sizes; (void)n_in; (void)out_size; (void)ws_size;
  const float* x    = (const float*)d_in[0];
  const float* ent  = (const float*)d_in[1];
  const float* mix  = (const float*)d_in[2];
  const float* rw   = (const float*)d_in[3];
  const float* encw = (const float*)d_in[4];
  const float* encb = (const float*)d_in[5];
  const float* decw = (const float*)d_in[6];
  const float* decb = (const float*)d_in[7];

  char* ws = (char*)d_ws;
  __bf16* At    = (__bf16*)(ws);
  __bf16* xgT   = (__bf16*)(ws + 134217728ull);
  float*  xb    = (float*)(ws + 142606336ull);
  float*  slot0 = (float*)(ws + 159383552ull);
  float*  slot1 = slot0 + (size_t)N_ * S_;
  float*  slot2 = slot1 + (size_t)N_ * S_;
  float*  slots[3] = {slot0, slot1, slot2};

  float* out   = (float*)d_out;                    // B*N*S
  float* nsout = out + (size_t)B_ * N_ * S_;       // N*S

  prep_mixing_kernel<<<dim3(N_ / 64, N_ / 64), 256, 0, stream>>>(mix, At);

  const float* cur = x;
  for (int l = 0; l < 3; ++l) {
    gate_xpose_kernel<<<dim3(N_ / 64, B_), 256, 0, stream>>>(cur, ent, xgT);
    gemm_mix_kernel<<<dim3(N_ / MT, B_), 256, 0, stream>>>(At, xgT, xb);
    mean_kernel<<<(N_ * S_) / 256, 256, 0, stream>>>(xb, slots[l]);
    cur = xb;
  }

  // temporal weights for T=3: RECENCY^{2,1,0} / sum
  const float wsum = 0.81f + 0.9f + 1.0f;
  final_ns_kernel<<<N_, 64, 0, stream>>>(slot0, slot1, slot2,
                                         encw, encb, decw, decb, nsout,
                                         0.81f / wsum, 0.9f / wsum, 1.0f / wsum);
  readout_kernel<<<(B_ * N_) / 4, 256, 0, stream>>>(xb, rw, out);
}
